// CrossStreamAttention_69887707840578
// MI455X (gfx1250) — matrix operand order repaired and checked by reading, not compiled, not run on hardware
//
#include <hip/hip_runtime.h>

// ---------------------------------------------------------------------------
// CDNA5 (gfx1250, wave32) cross-stream attention.
//   q = x_q @ Wq^T + bq  (scaled by 1/sqrt(D), stored bf16)
//   k = x_kv @ Wk^T + bk (bf16, row-major)
//   v = x_kv @ Wv^T + bv (bf16, stored TRANSPOSED [D][M] for contiguous B-frags)
//   out = softmax(q k^T) v   via flash-attention with v_wmma_f32_16x16x32_bf16
// K tiles are staged global->LDS with GLOBAL_LOAD_ASYNC_TO_LDS_B128 (ASYNCcnt
// double buffering) to overlap fetch with WMMA.
// ---------------------------------------------------------------------------

typedef __attribute__((ext_vector_type(16))) __bf16 v16bf;
typedef __attribute__((ext_vector_type(8)))  __bf16 v8bf;
typedef __attribute__((ext_vector_type(8)))  float  v8f;

#define DDIM 256
#define LOG2E 1.44269504088896340736f
#define KSTRIDE 40   // bf16 elements per staged K row (80B: bank-conflict-free)

__device__ __forceinline__ v8bf cvt8(const float4 a, const float4 b) {
  v8bf r;
  r[0] = (__bf16)a.x; r[1] = (__bf16)a.y; r[2] = (__bf16)a.z; r[3] = (__bf16)a.w;
  r[4] = (__bf16)b.x; r[5] = (__bf16)b.y; r[6] = (__bf16)b.z; r[7] = (__bf16)b.w;
  return r;
}

__device__ __forceinline__ v16bf cat16(const v8bf lo, const v8bf hi) {
  v16bf r;
#pragma unroll
  for (int e = 0; e < 8; ++e) { r[e] = lo[e]; r[8 + e] = hi[e]; }
  return r;
}

// wave32-safe XOR lane exchange via ds_bpermute (index is byte address of lane)
__device__ __forceinline__ float lane_xor_f32(float v, int mask) {
  const int lane = (int)(threadIdx.x & 31u);
  int r = __builtin_amdgcn_ds_bpermute(((lane ^ mask) << 2),
                                       __float_as_int(v));
  return __int_as_float(r);
}

// async global->LDS 16B copy per lane (tracked by ASYNCcnt)
__device__ __forceinline__ void async_copy_b128(unsigned lds_byte_addr,
                                                const void* gaddr) {
  asm volatile("global_load_async_to_lds_b128 %0, %1, off"
               :: "v"(lds_byte_addr),
                  "v"((unsigned long long)(uintptr_t)gaddr)
               : "memory");
}

// ---------------------------------------------------------------------------
// Projection: Y[r][c] (or Y^T) = sum_d X[r][d] * W[c][d] + b[c], one wave per
// 16x16 tile, K swept 32 at a time with WMMA bf16.
// ---------------------------------------------------------------------------
__global__ __launch_bounds__(128) void qkv_proj_kernel(
    const float* __restrict__ X, const float* __restrict__ W,
    const float* __restrict__ bias, __bf16* __restrict__ Y,
    int rows, float scale, int transposeOut)
{
  const int lane  = (int)(threadIdx.x & 31u);
  const int wid   = (int)blockIdx.x * 4 + (int)(threadIdx.x >> 5);
  const int tilesPerRow = DDIM / 16;                 // 16
  const int tileRow = wid / tilesPerRow;
  const int tileCol = wid % tilesPerRow;
  const int ln   = lane & 15;
  const int half = lane >> 4;
  const int kbaseA = half * 8;                       // A-frag K interleave base
  const int kbaseB = half * 16;                      // B-frag K base

  const size_t arow = (size_t)(tileRow * 16 + ln) * DDIM;
  const size_t brow = (size_t)(tileCol * 16 + ln) * DDIM;

  v8f acc = {};
#pragma unroll
  for (int s = 0; s < DDIM / 32; ++s) {
    const int k0 = s * 32;
    // A fragment (X rows, f32 -> bf16), interleaved chunks per ISA layout
    const float4 a0 = *reinterpret_cast<const float4*>(X + arow + k0 + kbaseA);
    const float4 a1 = *reinterpret_cast<const float4*>(X + arow + k0 + kbaseA + 4);
    const float4 a2 = *reinterpret_cast<const float4*>(X + arow + k0 + 16 + kbaseA);
    const float4 a3 = *reinterpret_cast<const float4*>(X + arow + k0 + 16 + kbaseA + 4);
    const v16bf af = cat16(cvt8(a0, a1), cvt8(a2, a3));
    // B fragment: row of W (y = x @ W^T), same interleaved K chunks as A
    const float4 b0 = *reinterpret_cast<const float4*>(W + brow + k0 + kbaseA);
    const float4 b1 = *reinterpret_cast<const float4*>(W + brow + k0 + kbaseA + 4);
    const float4 b2 = *reinterpret_cast<const float4*>(W + brow + k0 + 16 + kbaseA);
    const float4 b3 = *reinterpret_cast<const float4*>(W + brow + k0 + 16 + kbaseA + 4);
    const v16bf bf = cat16(cvt8(b0, b1), cvt8(b2, b3));
    acc = __builtin_amdgcn_wmma_f32_16x16x32_bf16(
        false, af, false, bf, (short)0, acc, false, false);
  }

  const int colg = tileCol * 16 + ln;
  const float bb = bias[colg];
#pragma unroll
  for (int r = 0; r < 8; ++r) {
    const int orow = tileRow * 16 + r + half * 8;
    const float v = (acc[r] + bb) * scale;
    if (transposeOut)
      Y[(size_t)colg * rows + orow] = (__bf16)v;    // V^T layout [D][rows]
    else
      Y[(size_t)orow * DDIM + colg] = (__bf16)v;    // row-major [rows][D]
  }
}

// ---------------------------------------------------------------------------
// Flash attention. Block = 4 waves, one 16-row q tile per block; the 4 waves
// split M into quarters (split-K over keys) and merge via LDS at the end.
// ---------------------------------------------------------------------------
__global__ __launch_bounds__(128) void flash_attn_kernel(
    const __bf16* __restrict__ qb, const __bf16* __restrict__ kb,
    const __bf16* __restrict__ vT, float* __restrict__ out,
    int N, int M)
{
  __shared__ __attribute__((aligned(16))) __bf16 kstage[4][2][32 * KSTRIDE];
  __shared__ __attribute__((aligned(16))) __bf16 pbuf[4][16 * 32]; // P transpose
  __shared__ float sob[4][8 * 256];                                // O merge (phase)
  __shared__ float sm[4][16], sl[4][16];                           // stats merge

  const int lane  = (int)(threadIdx.x & 31u);
  const int w     = (int)(threadIdx.x >> 5);
  const int qtile = (int)blockIdx.x;
  const int ln    = lane & 15;
  const int half  = lane >> 4;
  const int kbaseA = half * 8;
  const int kbaseB = half * 16;

  // per-lane piece of a K chunk copy: 128 segments of 16B over 4 instructions
  const int seg_row = lane >> 2;        // +8 per t
  const int seg_o16 = (lane & 3) * 8;   // element offset of 16B segment
  const unsigned lds_k0 = (unsigned)(uintptr_t)&kstage[w][0][0];
  const unsigned lds_k1 = (unsigned)(uintptr_t)&kstage[w][1][0];

  // ---- cache q fragments (16 rows x 256, bf16) in registers ----
  v16bf qf[8];
  const __bf16* qrowp = qb + (size_t)(qtile * 16 + ln) * DDIM;
#pragma unroll
  for (int s = 0; s < 8; ++s) {
    const int k0 = s * 32;
    const v8bf lo = *reinterpret_cast<const v8bf*>(qrowp + k0 + kbaseA);
    const v8bf hi = *reinterpret_cast<const v8bf*>(qrowp + k0 + 16 + kbaseA);
    qf[s] = cat16(lo, hi);
  }

  v8f o[16];
#pragma unroll
  for (int j = 0; j < 16; ++j) o[j] = (v8f){};
  float mrow[8], lrow[8];
#pragma unroll
  for (int r = 0; r < 8; ++r) { mrow[r] = -__builtin_inff(); lrow[r] = 0.f; }

  const int mPerWave = M / 4;
  const int mBeg = w * mPerWave;
  const int mEnd = mBeg + mPerWave;

  for (int m0 = mBeg; m0 < mEnd; m0 += 32) {
    __builtin_prefetch(vT + (size_t)ln * M + m0 + 32, 0, 1);

    // ---- prologue: async-stage K chunk s=0 into buffer 0 ----
    {
      const __bf16* g = kb + (size_t)m0 * DDIM;
#pragma unroll
      for (int t = 0; t < 4; ++t) {
        const int row = seg_row + t * 8;
        async_copy_b128(lds_k0 + (unsigned)(row * (KSTRIDE * 2) + seg_o16 * 2),
                        g + (size_t)row * DDIM + seg_o16);
      }
    }

    // ---- S = q k^T for 32 key columns (two 16-wide tiles) ----
    v8f s0 = (v8f){}, s1 = (v8f){};
#pragma unroll
    for (int s = 0; s < 8; ++s) {
      if (s < 7) {  // async-stage next K chunk into the other buffer
        const int k0n = (s + 1) * 32;
        const unsigned base = ((s + 1) & 1) ? lds_k1 : lds_k0;
        const __bf16* g = kb + (size_t)m0 * DDIM + k0n;
#pragma unroll
        for (int t = 0; t < 4; ++t) {
          const int row = seg_row + t * 8;
          async_copy_b128(base + (unsigned)(row * (KSTRIDE * 2) + seg_o16 * 2),
                          g + (size_t)row * DDIM + seg_o16);
        }
        asm volatile("s_wait_asynccnt 0x4" ::: "memory"); // current chunk done
      } else {
        asm volatile("s_wait_asynccnt 0x0" ::: "memory");
      }
      const __bf16* kc = &kstage[w][s & 1][0];
      const __bf16* kp0 = kc + ln * KSTRIDE + kbaseA;
      const v16bf kf0 = cat16(*(const v8bf*)kp0, *(const v8bf*)(kp0 + 16));
      s0 = __builtin_amdgcn_wmma_f32_16x16x32_bf16(
          false, qf[s], false, kf0, (short)0, s0, false, false);
      const __bf16* kp1 = kc + (16 + ln) * KSTRIDE + kbaseA;
      const v16bf kf1 = cat16(*(const v8bf*)kp1, *(const v8bf*)(kp1 + 16));
      s1 = __builtin_amdgcn_wmma_f32_16x16x32_bf16(
          false, qf[s], false, kf1, (short)0, s1, false, false);
    }

    // ---- online softmax (q was pre-scaled by 1/sqrt(D)) ----
    float alpha[8];
#pragma unroll
    for (int r = 0; r < 8; ++r) {
      float mx = fmaxf(s0[r], s1[r]);
      mx = fmaxf(mx, lane_xor_f32(mx, 1));
      mx = fmaxf(mx, lane_xor_f32(mx, 2));
      mx = fmaxf(mx, lane_xor_f32(mx, 4));
      mx = fmaxf(mx, lane_xor_f32(mx, 8));
      const float mnew = fmaxf(mrow[r], mx);
      alpha[r] = exp2f((mrow[r] - mnew) * LOG2E);
      const float p0 = exp2f((s0[r] - mnew) * LOG2E);
      const float p1 = exp2f((s1[r] - mnew) * LOG2E);
      float rs = p0 + p1;
      rs += lane_xor_f32(rs, 1);
      rs += lane_xor_f32(rs, 2);
      rs += lane_xor_f32(rs, 4);
      rs += lane_xor_f32(rs, 8);
      lrow[r] = lrow[r] * alpha[r] + rs;
      mrow[r] = mnew;
      const int prow = r + half * 8;
      pbuf[w][prow * 32 + ln]      = (__bf16)p0;   // D-layout -> LDS row-major
      pbuf[w][prow * 32 + 16 + ln] = (__bf16)p1;
    }
    asm volatile("s_wait_dscnt 0" ::: "memory");    // P stores visible to wave

    // ---- re-read P in A-fragment layout ----
    const __bf16* pp = &pbuf[w][ln * 32];
    const v16bf pf = cat16(*(const v8bf*)(pp + kbaseA),
                           *(const v8bf*)(pp + 16 + kbaseA));

    // ---- O = alpha*O + P V (16 output d-tiles) ----
#pragma unroll
    for (int j = 0; j < 16; ++j) {
      const __bf16* vp = vT + (size_t)(j * 16 + ln) * M + m0 + kbaseA;
      const v16bf vf = cat16(((const v8bf*)vp)[0], *(const v8bf*)(vp + 16));
      v8f oj = o[j];
#pragma unroll
      for (int r = 0; r < 8; ++r) oj[r] *= alpha[r];
      o[j] = __builtin_amdgcn_wmma_f32_16x16x32_bf16(
          false, pf, false, vf, (short)0, oj, false, false);
    }
  }

  // ---- cross-wave split-K merge through LDS ----
  if (ln == 0) {
#pragma unroll
    for (int r = 0; r < 8; ++r) {
      sm[w][r + half * 8] = mrow[r];
      sl[w][r + half * 8] = lrow[r];
    }
  }
  __syncthreads();

  for (int phase = 0; phase < 2; ++phase) {
#pragma unroll
    for (int jj = 0; jj < 8; ++jj) {
      const int j = phase * 8 + jj;
#pragma unroll
      for (int r = 0; r < 8; ++r)
        sob[w][jj * 256 + (r + half * 8) * 16 + ln] = o[j][r];
    }
    __syncthreads();
#pragma unroll
    for (int t = 0; t < 2; ++t) {
      const int j  = phase * 8 + w * 2 + t;
      const int jj = w * 2 + t;
#pragma unroll
      for (int r = 0; r < 8; ++r) {
        const int row = r + half * 8;
        const float gm = fmaxf(fmaxf(sm[0][row], sm[1][row]),
                               fmaxf(sm[2][row], sm[3][row]));
        float gl = 0.f, acc = 0.f;
#pragma unroll
        for (int ww = 0; ww < 4; ++ww) {
          const float wgt = exp2f((sm[ww][row] - gm) * LOG2E);
          gl  += sl[ww][row] * wgt;
          acc += sob[ww][jj * 256 + row * 16 + ln] * wgt;
        }
        out[(size_t)(qtile * 16 + row) * DDIM + j * 16 + ln] = acc / gl;
      }
    }
    __syncthreads();
  }
}

// ---------------------------------------------------------------------------
extern "C" void kernel_launch(void* const* d_in, const int* in_sizes, int n_in,
                              void* d_out, int out_size, void* d_ws, size_t ws_size,
                              hipStream_t stream) {
  (void)in_sizes; (void)n_in; (void)out_size;
  const int N = 8192, M = 8192;
  const float qscale = 0.0625f;  // 1/sqrt(256)

  const float* x_q  = (const float*)d_in[0];
  const float* x_kv = (const float*)d_in[1];
  const float* Wq   = (const float*)d_in[2];
  const float* bq   = (const float*)d_in[3];
  const float* Wk   = (const float*)d_in[4];
  const float* bk   = (const float*)d_in[5];
  const float* Wv   = (const float*)d_in[6];
  const float* bv   = (const float*)d_in[7];

  const size_t elems = (size_t)N * DDIM;            // == M*DDIM
  if (ws_size < 3u * elems * sizeof(__bf16)) return;
  __bf16* qb = (__bf16*)d_ws;
  __bf16* kb = qb + elems;
  __bf16* vT = kb + elems;

  const int projBlocks = (N / 16) * (DDIM / 16) / 4;  // 2048
  qkv_proj_kernel<<<projBlocks, 128, 0, stream>>>(x_q,  Wq, bq, qb, N, qscale, 0);
  qkv_proj_kernel<<<projBlocks, 128, 0, stream>>>(x_kv, Wk, bk, kb, M, 1.0f,   0);
  qkv_proj_kernel<<<projBlocks, 128, 0, stream>>>(x_kv, Wv, bv, vT, M, 1.0f,   1);

  flash_attn_kernel<<<N / 16, 128, 0, stream>>>(qb, kb, vT, (float*)d_out, N, M);
}
